// MoELayer_5712306504199
// MI455X (gfx1250) — compile-verified, matrix operand order checked by reading, not run on hardware
//
#include <hip/hip_runtime.h>
#include <hip/hip_bf16.h>
#include <math.h>

// Problem constants (from reference): B=4, S=1024, D=1024, E=8, FF=4096, K=2
#define T_TOK   4096      // B*S tokens
#define D_DIM   1024
#define E_EXP   8
#define FF_DIM  4096
#define TOPK    2
#define BM      128       // GEMM tile M (slots)
#define BN      128       // GEMM tile N
#define BK      32        // GEMM tile K (matches wmma_f32_16x16x32_bf16)
#define NSLOT   9216      // T*K + E*(BM-1) rounded up to BM multiple (72 tiles)
#define NTHREADS 256      // 8 waves (wave32)

typedef __bf16 bf16;
typedef __attribute__((ext_vector_type(16))) __bf16 bf16x16;
typedef __attribute__((ext_vector_type(8)))  float  f32x8;

union Frag16 { bf16x16 v; uint4 q[2]; };

// CDNA5 async global->LDS path (guarded; fallback = plain loads).
// Probe round 1 showed the builtin expects:
//   (int __vector_size(16) AS1*, int __vector_size(16) AS3*, imm int, imm int)
#if defined(__gfx1250__) && __has_builtin(__builtin_amdgcn_global_load_async_to_lds_b128) && __has_builtin(__builtin_amdgcn_s_wait_asynccnt)
#define MOE_ASYNC_LDS 1
typedef int v4i_gcc __attribute__((vector_size(16)));
typedef __attribute__((address_space(1))) v4i_gcc as1_v4i;
typedef __attribute__((address_space(3))) v4i_gcc as3_v4i;
#else
#define MOE_ASYNC_LDS 0
#endif

__device__ __forceinline__ unsigned pack2(float a, float b) {
  union { __bf16 h[2]; unsigned u; } p;
  p.h[0] = (__bf16)a; p.h[1] = (__bf16)b;
  return p.u;
}

// ---------------------------------------------------------------- convert x
__global__ void k_cvt(const float* __restrict__ x, bf16* __restrict__ xbf) {
  int i4 = (blockIdx.x * blockDim.x + threadIdx.x) * 4;
  if (i4 + 3 < T_TOK * D_DIM) {
    float4 v = *(const float4*)(x + i4);
    xbf[i4 + 0] = (bf16)v.x; xbf[i4 + 1] = (bf16)v.y;
    xbf[i4 + 2] = (bf16)v.z; xbf[i4 + 3] = (bf16)v.w;
  }
}

// ---------------------------------------------------------------- gating
__global__ void k_gate(const float* __restrict__ x, const float* __restrict__ gw,
                       const float* __restrict__ gb, int* __restrict__ tok_eidx,
                       float* __restrict__ tok_w) {
  int t = blockIdx.x * blockDim.x + threadIdx.x;
  if (t >= T_TOK) return;
  float acc[E_EXP];
#pragma unroll
  for (int e = 0; e < E_EXP; ++e) acc[e] = gb[e];
  const float* xr = x + (size_t)t * D_DIM;
  for (int d = 0; d < D_DIM; ++d) {
    float xv = xr[d];
    const float* g = gw + (size_t)d * E_EXP;
#pragma unroll
    for (int e = 0; e < E_EXP; ++e) acc[e] = fmaf(xv, g[e], acc[e]);
  }
  // top-2, lowest index wins ties (matches jax.lax.top_k)
  int i0 = 0;
#pragma unroll
  for (int e = 1; e < E_EXP; ++e) if (acc[e] > acc[i0]) i0 = e;
  int i1 = (i0 == 0) ? 1 : 0;
#pragma unroll
  for (int e = 0; e < E_EXP; ++e) if (e != i0 && acc[e] > acc[i1]) i1 = e;
  float m  = fmaxf(acc[i0], acc[i1]);
  float e0 = expf(acc[i0] - m), e1 = expf(acc[i1] - m);
  float inv = 1.0f / (e0 + e1);
  tok_eidx[2 * t]     = i0;  tok_eidx[2 * t + 1] = i1;
  tok_w[2 * t]        = e0 * inv;
  tok_w[2 * t + 1]    = e1 * inv;
}

// ------------------------------------------------- deterministic routing
__global__ void k_route(const int* __restrict__ tok_eidx,
                        int* __restrict__ offs, int* __restrict__ tok_slot,
                        int* __restrict__ slot_tok) {
  __shared__ int s_cnt[E_EXP];
  __shared__ int s_off[E_EXP + 1];
  const int tid = threadIdx.x;
  for (int i = tid; i < NSLOT; i += NTHREADS) slot_tok[i] = -1;
  if (tid < E_EXP) {
    int c = 0;
    for (int i = 0; i < 2 * T_TOK; ++i) c += (tok_eidx[i] == tid) ? 1 : 0;
    s_cnt[tid] = c;
  }
  __syncthreads();
  if (tid == 0) {
    int off = 0;
    s_off[0] = 0;
    for (int e = 0; e < E_EXP; ++e) {
      off += ((s_cnt[e] + BM - 1) / BM) * BM;   // pad: tiles never span experts
      s_off[e + 1] = off;
    }
    for (int e = 0; e <= E_EXP; ++e) offs[e] = s_off[e];
  }
  __syncthreads();
  if (tid < E_EXP) {                            // token-order assignment: deterministic
    int pos = s_off[tid];
    for (int i = 0; i < 2 * T_TOK; ++i) {
      if (tok_eidx[i] == tid) {
        slot_tok[pos] = i >> 1;
        tok_slot[i]   = pos;
        ++pos;
      }
    }
  }
}

// ------------------------------------------- GEMM1: h = gelu(x @ w1 + b1)
__global__ __launch_bounds__(NTHREADS) void k_ffn1(
    const bf16* __restrict__ xbf, const float* __restrict__ w1,
    const float* __restrict__ b1, bf16* __restrict__ hbuf,
    const int* __restrict__ slot_tok, const int* __restrict__ offs) {
  const int tid = threadIdx.x;
  const int slotbase = blockIdx.y * BM;
  if (slotbase >= offs[E_EXP]) return;
  int e = 0;
#pragma unroll
  for (int i = 1; i <= E_EXP; ++i) if (slotbase >= offs[i]) e = i;
  const float* wE = w1 + (size_t)e * D_DIM * FF_DIM;
  const int nbase = blockIdx.x * BN;

  __shared__ bf16 sA[BM][BK];   // row-major, 64B rows
  __shared__ bf16 sB[BN][BK];   // [col][k]: each lane's frag contiguous

  const int lane = tid & 31, wv = tid >> 5;
  const int waveM = wv >> 2, waveN = wv & 3;    // 2x4 wave grid -> 64x32 per wave
  const int l16 = lane & 15, half = lane >> 4;

  f32x8 zero = {};
  f32x8 acc[4][2];
#pragma unroll
  for (int i = 0; i < 4; ++i)
#pragma unroll
    for (int j = 0; j < 2; ++j) acc[i][j] = zero;

  const int lr = tid >> 1;                 // A loader: row 0..127
  const int lc16 = (tid & 1) * 16;         // 32B half-row
  const int ltok = slot_tok[slotbase + lr];
  const int kp = tid >> 4;                 // B loader: k-pair 0..15
  const int c0 = (tid & 15) * 8;           // 8 columns

  for (int kk = 0; kk < D_DIM; kk += BK) {
    // A tile: gather token rows (bf16, already converted)
    uint4 v0 = make_uint4(0, 0, 0, 0), v1 = v0;
    if (ltok >= 0) {
      const uint4* g = (const uint4*)(xbf + (size_t)ltok * D_DIM + kk + lc16);
      v0 = g[0]; v1 = g[1];
    }
    *(uint4*)&sA[lr][lc16]     = v0;
    *(uint4*)&sA[lr][lc16 + 8] = v1;
    // B tile: f32 weights -> bf16, transposed into [col][k]
    {
      const float* g0 = wE + (size_t)(kk + 2 * kp) * FF_DIM + nbase + c0;
      const float* g1 = g0 + FF_DIM;
      float4 a0 = *(const float4*)g0, a1 = *(const float4*)(g0 + 4);
      float4 c0v = *(const float4*)g1, c1v = *(const float4*)(g1 + 4);
      float t0[8] = {a0.x, a0.y, a0.z, a0.w, a1.x, a1.y, a1.z, a1.w};
      float t1[8] = {c0v.x, c0v.y, c0v.z, c0v.w, c1v.x, c1v.y, c1v.z, c1v.w};
#pragma unroll
      for (int cc = 0; cc < 8; ++cc)
        *(unsigned*)&sB[c0 + cc][2 * kp] = pack2(t0[cc], t1[cc]);
      if (kk + BK < D_DIM) __builtin_prefetch(g0 + (size_t)BK * FF_DIM, 0, 0);
    }
    __syncthreads();
    Frag16 af[4], bfr[2];
#pragma unroll
    for (int i = 0; i < 4; ++i) {
      const bf16* p = &sA[waveM * 64 + i * 16 + l16][half * 8];
      af[i].q[0] = *(const uint4*)p;
      af[i].q[1] = *(const uint4*)(p + 16);
    }
#pragma unroll
    for (int j = 0; j < 2; ++j) {
      const bf16* p = &sB[waveN * 32 + j * 16 + l16][half * 8];
      bfr[j].q[0] = *(const uint4*)p;
      bfr[j].q[1] = *(const uint4*)(p + 16);
    }
#pragma unroll
    for (int i = 0; i < 4; ++i)
#pragma unroll
      for (int j = 0; j < 2; ++j)
        acc[i][j] = __builtin_amdgcn_wmma_f32_16x16x32_bf16(
            false, af[i].v, false, bfr[j].v, (short)0, acc[i][j], false, false);
    __syncthreads();
  }
  // epilogue: + b1, exact erf-GELU, store bf16
#pragma unroll
  for (int i = 0; i < 4; ++i) {
#pragma unroll
    for (int j = 0; j < 2; ++j) {
      const int colg = nbase + waveN * 32 + j * 16 + l16;
      const float bias = b1[(size_t)e * FF_DIM + colg];
      const int row0 = slotbase + waveM * 64 + i * 16 + half * 8;
#pragma unroll
      for (int k = 0; k < 8; ++k) {
        float v = acc[i][j][k] + bias;
        v = 0.5f * v * (1.0f + erff(v * 0.70710678118654752440f));
        hbuf[(size_t)(row0 + k) * FF_DIM + colg] = (bf16)v;
      }
    }
  }
}

// ------------------------------------------- GEMM2: y = h @ w2 + b2
__global__ __launch_bounds__(NTHREADS) void k_ffn2(
    const bf16* __restrict__ hbuf, const float* __restrict__ w2,
    const float* __restrict__ b2, float* __restrict__ ybuf,
    const int* __restrict__ offs) {
  const int tid = threadIdx.x;
  const int slotbase = blockIdx.y * BM;
  if (slotbase >= offs[E_EXP]) return;
  int e = 0;
#pragma unroll
  for (int i = 1; i <= E_EXP; ++i) if (slotbase >= offs[i]) e = i;
  const float* wE = w2 + (size_t)e * FF_DIM * D_DIM;
  const int nbase = blockIdx.x * BN;

  __shared__ bf16 sA[BM][BK];
  __shared__ bf16 sB[BN][BK];

  const int lane = tid & 31, wv = tid >> 5;
  const int waveM = wv >> 2, waveN = wv & 3;
  const int l16 = lane & 15, half = lane >> 4;

  f32x8 zero = {};
  f32x8 acc[4][2];
#pragma unroll
  for (int i = 0; i < 4; ++i)
#pragma unroll
    for (int j = 0; j < 2; ++j) acc[i][j] = zero;

  const int lr = tid >> 1;
  const int lc16 = (tid & 1) * 16;
  const int kp = tid >> 4;
  const int c0 = (tid & 15) * 8;

  for (int kk = 0; kk < FF_DIM; kk += BK) {
    // A tile: contiguous bf16 slot rows of h — CDNA5 async global->LDS if available
#if MOE_ASYNC_LDS
    {
      const bf16* g = hbuf + (size_t)(slotbase + lr) * FF_DIM + kk + lc16;
      bf16* l = &sA[lr][lc16];
      __builtin_amdgcn_global_load_async_to_lds_b128(
          (as1_v4i*)(uintptr_t)g, (as3_v4i*)(uintptr_t)l, 0, 0);
      __builtin_amdgcn_global_load_async_to_lds_b128(
          (as1_v4i*)(uintptr_t)(g + 8), (as3_v4i*)(uintptr_t)(l + 8), 0, 0);
    }
#else
    {
      const uint4* g = (const uint4*)(hbuf + (size_t)(slotbase + lr) * FF_DIM + kk + lc16);
      *(uint4*)&sA[lr][lc16]     = g[0];
      *(uint4*)&sA[lr][lc16 + 8] = g[1];
    }
#endif
    // B tile: f32 w2 -> bf16 transposed
    {
      const float* g0 = wE + (size_t)(kk + 2 * kp) * D_DIM + nbase + c0;
      const float* g1 = g0 + D_DIM;
      float4 a0 = *(const float4*)g0, a1 = *(const float4*)(g0 + 4);
      float4 c0v = *(const float4*)g1, c1v = *(const float4*)(g1 + 4);
      float t0[8] = {a0.x, a0.y, a0.z, a0.w, a1.x, a1.y, a1.z, a1.w};
      float t1[8] = {c0v.x, c0v.y, c0v.z, c0v.w, c1v.x, c1v.y, c1v.z, c1v.w};
#pragma unroll
      for (int cc = 0; cc < 8; ++cc)
        *(unsigned*)&sB[c0 + cc][2 * kp] = pack2(t0[cc], t1[cc]);
      if (kk + BK < FF_DIM) __builtin_prefetch(g0 + (size_t)BK * D_DIM, 0, 0);
    }
#if MOE_ASYNC_LDS
    __builtin_amdgcn_s_wait_asynccnt(0);
#endif
    __syncthreads();
    Frag16 af[4], bfr[2];
#pragma unroll
    for (int i = 0; i < 4; ++i) {
      const bf16* p = &sA[waveM * 64 + i * 16 + l16][half * 8];
      af[i].q[0] = *(const uint4*)p;
      af[i].q[1] = *(const uint4*)(p + 16);
    }
#pragma unroll
    for (int j = 0; j < 2; ++j) {
      const bf16* p = &sB[waveN * 32 + j * 16 + l16][half * 8];
      bfr[j].q[0] = *(const uint4*)p;
      bfr[j].q[1] = *(const uint4*)(p + 16);
    }
#pragma unroll
    for (int i = 0; i < 4; ++i)
#pragma unroll
      for (int j = 0; j < 2; ++j)
        acc[i][j] = __builtin_amdgcn_wmma_f32_16x16x32_bf16(
            false, af[i].v, false, bfr[j].v, (short)0, acc[i][j], false, false);
    __syncthreads();
  }
#pragma unroll
  for (int i = 0; i < 4; ++i) {
#pragma unroll
    for (int j = 0; j < 2; ++j) {
      const int colg = nbase + waveN * 32 + j * 16 + l16;
      const float bias = b2[(size_t)e * D_DIM + colg];
      const int row0 = slotbase + waveM * 64 + i * 16 + half * 8;
#pragma unroll
      for (int k = 0; k < 8; ++k)
        ybuf[(size_t)(row0 + k) * D_DIM + colg] = acc[i][j][k] + bias;
    }
  }
}

// ---------------------------------------------------------------- combine
__global__ void k_combine(const float* __restrict__ ybuf,
                          const int* __restrict__ tok_slot,
                          const float* __restrict__ tok_w,
                          float* __restrict__ out) {
  size_t idx = (size_t)blockIdx.x * blockDim.x + threadIdx.x;
  if (idx >= (size_t)T_TOK * D_DIM) return;
  int t = (int)(idx >> 10);
  int d = (int)(idx & 1023);
  int s0 = tok_slot[2 * t], s1 = tok_slot[2 * t + 1];
  float w0 = tok_w[2 * t], w1 = tok_w[2 * t + 1];
  out[idx] = w0 * ybuf[(size_t)s0 * D_DIM + d] + w1 * ybuf[(size_t)s1 * D_DIM + d];
}

// ---------------------------------------------------------------- launch
extern "C" void kernel_launch(void* const* d_in, const int* in_sizes, int n_in,
                              void* d_out, int out_size, void* d_ws, size_t ws_size,
                              hipStream_t stream) {
  const float* x  = (const float*)d_in[0];
  const float* gw = (const float*)d_in[1];
  const float* gb = (const float*)d_in[2];
  const float* w1 = (const float*)d_in[3];
  const float* b1 = (const float*)d_in[4];
  const float* w2 = (const float*)d_in[5];
  const float* b2 = (const float*)d_in[6];
  float* out = (float*)d_out;

  char* ws = (char*)d_ws;
  // metadata region (first 1 MB)
  int*   offs     = (int*)(ws);                                   // 9 ints
  int*   tok_eidx = (int*)(ws + 256);                             // 2T ints
  float* tok_w    = (float*)(ws + 256 + (size_t)2 * T_TOK * 4);   // 2T floats
  int*   tok_slot = (int*)(ws + 256 + (size_t)4 * T_TOK * 4);     // 2T ints
  int*   slot_tok = (int*)(ws + 256 + (size_t)6 * T_TOK * 4);     // NSLOT ints
  // bulk buffers: xbf16 8MB, h bf16 72MB, y f32 36MB  (total ~122MB)
  bf16*  xbf  = (bf16*)(ws + (size_t)(1 << 20));
  bf16*  hbuf = (bf16*)(ws + (size_t)(1 << 20) + (size_t)T_TOK * D_DIM * 2);
  float* ybuf = (float*)(ws + (size_t)(1 << 20) + (size_t)T_TOK * D_DIM * 2
                            + (size_t)NSLOT * FF_DIM * 2);

  k_cvt<<<(T_TOK * D_DIM / 4 + NTHREADS - 1) / NTHREADS, NTHREADS, 0, stream>>>(x, xbf);
  k_gate<<<(T_TOK + NTHREADS - 1) / NTHREADS, NTHREADS, 0, stream>>>(x, gw, gb, tok_eidx, tok_w);
  k_route<<<1, NTHREADS, 0, stream>>>(tok_eidx, offs, tok_slot, slot_tok);
  k_ffn1<<<dim3(FF_DIM / BN, NSLOT / BM), NTHREADS, 0, stream>>>(xbf, w1, b1, hbuf, slot_tok, offs);
  k_ffn2<<<dim3(D_DIM / BN, NSLOT / BM), NTHREADS, 0, stream>>>(hbuf, w2, b2, ybuf, offs);
  k_combine<<<(T_TOK * D_DIM + NTHREADS - 1) / NTHREADS, NTHREADS, 0, stream>>>(ybuf, tok_slot, tok_w, out);
}
